// GGNet_11012296147741
// MI455X (gfx1250) — matrix operand order repaired and checked by reading, not compiled, not run on hardware
//
#include <hip/hip_runtime.h>
#include <hip/hip_bf16.h>
#include <stdint.h>

// ---------------- problem constants (match reference) ----------------
#define HD       64      // H
#define NDIM     133     // node feature dim
#define BDIM     16      // bond feature dim
#define INTD     128     // internal dim of edge MLP
#define NGRAPH   64      // graphs per batch (B)
#define DEPTH    3
#define STEPS    3

typedef _Float16 v16h __attribute__((ext_vector_type(16)));
typedef float    v8f  __attribute__((ext_vector_type(8)));

struct __align__(4) h2 { _Float16 x, y; };

__device__ __forceinline__ float sigf(float x) { return 1.0f / (1.0f + __expf(-x)); }

__device__ __forceinline__ void atomAddF(float* p, float v) {
  __hip_atomic_fetch_add(p, v, __ATOMIC_RELAXED, __HIP_MEMORY_SCOPE_AGENT);
}

// order-preserving float <-> uint encoding for atomicMax-based segment max
__device__ __forceinline__ unsigned ford(float f) {
  unsigned u = __float_as_uint(f);
  return (u & 0x80000000u) ? ~u : (u | 0x80000000u);
}
__device__ __forceinline__ float funord(unsigned o) {
  unsigned u = (o & 0x80000000u) ? (o & 0x7fffffffu) : ~o;
  return __uint_as_float(u);
}

// ---------------- generic zero ----------------
__global__ void k_zero(float* p, int n) {
  int i = blockIdx.x * blockDim.x + threadIdx.x;
  if (i < n) p[i] = 0.0f;
}

// ---------------- out = relu(x @ W_expand + b)  (out == h) ----------------
__global__ void k_expand(const float* __restrict__ x, const float* __restrict__ W,
                         const float* __restrict__ b, float* __restrict__ out, int N) {
  int n = blockIdx.x;
  __shared__ float xr[NDIM];
  for (int i = threadIdx.x; i < NDIM; i += HD) xr[i] = x[(size_t)n * NDIM + i];
  __syncthreads();
  int t = threadIdx.x;
  float acc = b[t];
  #pragma unroll 7
  for (int k = 0; k < NDIM; ++k) acc += xr[k] * W[k * HD + t];
  out[(size_t)n * HD + t] = fmaxf(acc, 0.0f);
}

// ---------------- cnt[dst] += 1 ----------------
__global__ void k_count(const int* __restrict__ ei, float* __restrict__ cnt, int E) {
  int e = blockIdx.x * blockDim.x + threadIdx.x;
  if (e < E) atomAddF(&cnt[ei[E + e]], 1.0f);
}

// ---------------- A = relu(edge_attr @ We1 + be1) -> f16 [E,128] ----------------
__global__ void k_edge_mlp(const float* __restrict__ ea, const float* __restrict__ We1,
                           const float* __restrict__ be1, _Float16* __restrict__ A, int E) {
  int e = blockIdx.x;
  __shared__ float er[BDIM];
  int t = threadIdx.x;  // 128
  if (t < BDIM) er[t] = ea[(size_t)e * BDIM + t];
  __syncthreads();
  float acc = be1[t];
  #pragma unroll
  for (int k = 0; k < BDIM; ++k) acc += er[k] * We1[k * INTD + t];
  A[(size_t)e * INTD + t] = (_Float16)fmaxf(acc, 0.0f);
}

// ---------------- We2 f32 -> f16 ----------------
__global__ void k_cvt(const float* __restrict__ src, _Float16* __restrict__ dst, int n) {
  int i = blockIdx.x * blockDim.x + threadIdx.x;
  if (i < n) dst[i] = (_Float16)src[i];
}

// ---------------- big WMMA GEMM: ew[E,4096]f16 = A[E,128]f16 @ We2h[128,4096]f16 + be2 ----
#define TSTR 136   // padded LDS row stride in halves (272B: 16B-aligned rows, bank spread)

__device__ __forceinline__ v16h frag_a(const _Float16* Ah, int row, int k0, int grp) {
  // ISA 16-bit A 16x32 layout: lanes0-15 K{0..7,16..23}; lanes16-31 K{8..15,24..31}
  v16h a;
  const _Float16* p = Ah + row * TSTR + k0;
  #pragma unroll
  for (int j = 0; j < 8; ++j) {
    int koff = ((j < 4) ? 2 * j : 2 * j + 8) + (grp ? 8 : 0);
    h2 v = *(const h2*)(p + koff);
    a[2 * j] = v.x; a[2 * j + 1] = v.y;
  }
  return a;
}
__device__ __forceinline__ v16h frag_b(const _Float16* Bt, int ncol, int k0, int grp) {
  // ISA 16-bit B 32x16 layout: lane = n; lanes0-15 K{0..7,16..23}; lanes16-31 K{8..15,24..31}
  v16h b;
  const _Float16* p = Bt + ncol * TSTR + k0 + (grp ? 8 : 0);
  #pragma unroll
  for (int j = 0; j < 8; ++j) {
    h2 v = *(const h2*)(p + ((j < 4) ? 2 * j : 2 * j + 8));
    b[2 * j] = v.x; b[2 * j + 1] = v.y;
  }
  return b;
}

__global__ __launch_bounds__(256)
void k_ew_wmma(const _Float16* __restrict__ Ag, const _Float16* __restrict__ Bg,
               const float* __restrict__ be2, _Float16* __restrict__ ewg, int E) {
  __shared__ __align__(16) _Float16 Ah[128 * TSTR];  // 128 edges x 128 K
  __shared__ __align__(16) _Float16 Bt[64 * TSTR];   // transposed: [n][k], 64 n x 128 k
  const int e0 = blockIdx.x * 128;
  const int n0 = blockIdx.y * 64;

  // stage A tile: 128 rows x 64 halves, 16B chunks
  for (int i = threadIdx.x; i < 1024; i += 256) {
    int r = i >> 3, c = (i & 7) * 8;
    int e = e0 + r;
    uint4 v = make_uint4(0u, 0u, 0u, 0u);
    if (e < E) v = *(const uint4*)(Ag + (size_t)e * INTD + c);
    *(uint4*)(&Ah[r * TSTR + c]) = v;
  }
  // stage B tile transposed: read We2h[k][n0+..] coalesced, scatter to Bt[n][k]
  for (int i = threadIdx.x; i < 4096; i += 256) {
    int k = i >> 5, n2 = (i & 31);
    h2 v = *(const h2*)(Bg + (size_t)k * 4096 + n0 + n2 * 2);
    Bt[(n2 * 2) * TSTR + k]     = v.x;
    Bt[(n2 * 2 + 1) * TSTR + k] = v.y;
  }
  __syncthreads();

  const int w = threadIdx.x >> 5;
  const int lane = threadIdx.x & 31;
  const int grp = lane >> 4;
  const int l16 = lane & 15;

  v8f acc[4] = {};
  #pragma unroll
  for (int kb = 0; kb < 4; ++kb) {
    v16h a = frag_a(Ah, w * 16 + l16, kb * 32, grp);
    #pragma unroll
    for (int nt = 0; nt < 4; ++nt) {
      v16h b = frag_b(Bt, nt * 16 + l16, kb * 32, grp);
      acc[nt] = __builtin_amdgcn_wmma_f32_16x16x32_f16(
          false, a, false, b, (short)0, acc[nt], false, false);
    }
  }

  // C layout: VGPR r -> row (r | grp*8), col = lane&15
  #pragma unroll
  for (int nt = 0; nt < 4; ++nt) {
    int n = n0 + nt * 16 + l16;
    float bias = be2[n];
    #pragma unroll
    for (int r = 0; r < 8; ++r) {
      int e = e0 + w * 16 + r + (grp ? 8 : 0);
      if (e < E) ewg[(size_t)e * 4096 + n] = (_Float16)(acc[nt][r] + bias);
    }
  }
}

// ---------------- per-edge matvec + scatter: aggr[dst] += out[src] @ ew[e] ----------------
__global__ void k_msg(const int* __restrict__ ei, const _Float16* __restrict__ ew,
                      const float* __restrict__ out, float* __restrict__ aggr, int E) {
  int e = blockIdx.x * 8 + (threadIdx.x >> 5);
  int lane = threadIdx.x & 31;
  if (e >= E) return;
  int src = ei[e], dst = ei[E + e];
  float s0 = out[(size_t)src * HD + lane];
  float s1 = out[(size_t)src * HD + 32 + lane];
  float acc0 = 0.0f, acc1 = 0.0f;
  const _Float16* ewe = ew + (size_t)e * 4096;
  #pragma unroll 4
  for (int h = 0; h < HD; ++h) {
    float s = (h < 32) ? __shfl(s0, h, 32) : __shfl(s1, h - 32, 32);
    acc0 += s * (float)ewe[h * HD + lane];
    acc1 += s * (float)ewe[h * HD + 32 + lane];
  }
  atomAddF(&aggr[(size_t)dst * HD + lane], acc0);
  atomAddF(&aggr[(size_t)dst * HD + 32 + lane], acc1);
}

// ---------------- NNConv root + mean aggr + GRU step (out == h, in place) --------------
__global__ void k_node(float* __restrict__ out, const float* __restrict__ aggr,
                       const float* __restrict__ cnt, const float* __restrict__ Wroot,
                       const float* __restrict__ bconv, const float* __restrict__ Wih,
                       const float* __restrict__ Whh, const float* __restrict__ bih,
                       const float* __restrict__ bhh, int N) {
  int n = blockIdx.x, t = threadIdx.x;
  __shared__ float orow[HD], mrow[HD];
  orow[t] = out[(size_t)n * HD + t];
  __syncthreads();
  float c = fmaxf(cnt[n], 1.0f);
  float acc = aggr[(size_t)n * HD + t] / c + bconv[t];
  #pragma unroll 8
  for (int k = 0; k < HD; ++k) acc += orow[k] * Wroot[k * HD + t];
  mrow[t] = fmaxf(acc, 0.0f);
  __syncthreads();
  float gir = bih[t], giz = bih[HD + t], gin = bih[2 * HD + t];
  float ghr = bhh[t], ghz = bhh[HD + t], ghn = bhh[2 * HD + t];
  #pragma unroll 4
  for (int k = 0; k < HD; ++k) {
    float m = mrow[k], h = orow[k];
    gir += m * Wih[t * HD + k];            ghr += h * Whh[t * HD + k];
    giz += m * Wih[(HD + t) * HD + k];     ghz += h * Whh[(HD + t) * HD + k];
    gin += m * Wih[(2 * HD + t) * HD + k]; ghn += h * Whh[(2 * HD + t) * HD + k];
  }
  float r = sigf(gir + ghr);
  float z = sigf(giz + ghz);
  float nn = tanhf(gin + r * ghn);
  out[(size_t)n * HD + t] = (1.0f - z) * nn + z * orow[t];
}

// ---------------- Set2Set: one LSTM step (hl double-buffered) ----------------
__global__ void k_lstm(const float* __restrict__ qstar, const float* __restrict__ hl_in,
                       float* __restrict__ cl, float* __restrict__ hl_out,
                       const float* __restrict__ Wih, const float* __restrict__ Whh,
                       const float* __restrict__ bih, const float* __restrict__ bhh) {
  int idx = blockIdx.x * blockDim.x + threadIdx.x;  // 4096 = 64 graphs * 64
  int b = idx >> 6, t = idx & 63;
  float gi = bih[t] + bhh[t];
  float gf = bih[HD + t] + bhh[HD + t];
  float gg = bih[2 * HD + t] + bhh[2 * HD + t];
  float go = bih[3 * HD + t] + bhh[3 * HD + t];
  const float* q = qstar + b * 2 * HD;
  #pragma unroll 4
  for (int k = 0; k < 2 * HD; ++k) {
    float v = q[k];
    gi += v * Wih[t * 2 * HD + k];
    gf += v * Wih[(HD + t) * 2 * HD + k];
    gg += v * Wih[(2 * HD + t) * 2 * HD + k];
    go += v * Wih[(3 * HD + t) * 2 * HD + k];
  }
  const float* hh = hl_in + b * HD;
  #pragma unroll 4
  for (int k = 0; k < HD; ++k) {
    float v = hh[k];
    gi += v * Whh[t * HD + k];
    gf += v * Whh[(HD + t) * HD + k];
    gg += v * Whh[(2 * HD + t) * HD + k];
    go += v * Whh[(3 * HD + t) * HD + k];
  }
  float cn = sigf(gf) * cl[idx] + sigf(gi) * tanhf(gg);
  cl[idx] = cn;
  hl_out[idx] = sigf(go) * tanhf(cn);
}

// ---------------- attention logits + segment max ----------------
__global__ void k_logits(const float* __restrict__ out, const float* __restrict__ hl,
                         const int* __restrict__ batch, float* __restrict__ e_ws,
                         unsigned* __restrict__ emax, int N) {
  int n = blockIdx.x * blockDim.x + threadIdx.x;
  if (n >= N) return;
  int b = batch[n];
  const float* q = hl + b * HD;
  float e = 0.0f;
  #pragma unroll 8
  for (int k = 0; k < HD; ++k) e += out[(size_t)n * HD + k] * q[k];
  e_ws[n] = e;
  atomicMax(&emax[b], ford(e));
}

// ---------------- softmax numerator + segment sum ----------------
__global__ void k_attn(const float* __restrict__ e_ws, const unsigned* __restrict__ emax,
                       const int* __restrict__ batch, float* __restrict__ a_ws,
                       float* __restrict__ asum, int N) {
  int n = blockIdx.x * blockDim.x + threadIdx.x;
  if (n >= N) return;
  int b = batch[n];
  float a = __expf(e_ws[n] - funord(emax[b]));
  a_ws[n] = a;
  atomAddF(&asum[b], a);
}

// ---------------- weighted readout r_read[b] += (a/asum) * out[n] ----------------
__global__ void k_read(const float* __restrict__ a_ws, const float* __restrict__ asum,
                       const int* __restrict__ batch, const float* __restrict__ out,
                       float* __restrict__ r_read, int NT) {
  int idx = blockIdx.x * blockDim.x + threadIdx.x;
  if (idx >= NT) return;
  int n = idx >> 6, t = idx & 63;
  int b = batch[n];
  float w = a_ws[n] / asum[b];
  atomAddF(&r_read[b * HD + t], w * out[idx]);
}

// ---------------- q_star = concat(hl, r_read) ----------------
__global__ void k_qstar(const float* __restrict__ hl, const float* __restrict__ r_read,
                        float* __restrict__ qstar) {
  int idx = blockIdx.x * blockDim.x + threadIdx.x;  // 8192
  int b = idx >> 7, t = idx & 127;
  qstar[idx] = (t < HD) ? hl[b * HD + t] : r_read[b * HD + (t - HD)];
}

__global__ void k_copy(const float* __restrict__ s, float* __restrict__ d, int n) {
  int i = blockIdx.x * blockDim.x + threadIdx.x;
  if (i < n) d[i] = s[i];
}

// =======================================================================
extern "C" void kernel_launch(void* const* d_in, const int* in_sizes, int n_in,
                              void* d_out, int out_size, void* d_ws, size_t ws_size,
                              hipStream_t stream) {
  const float* x        = (const float*)d_in[0];
  const int*   ei       = (const int*)d_in[1];
  const float* ea       = (const float*)d_in[2];
  const int*   batch    = (const int*)d_in[3];
  const float* W_expand = (const float*)d_in[4];
  const float* b_expand = (const float*)d_in[5];
  const float* We1      = (const float*)d_in[6];
  const float* be1      = (const float*)d_in[7];
  const float* We2      = (const float*)d_in[8];
  const float* be2      = (const float*)d_in[9];
  const float* W_root   = (const float*)d_in[10];
  const float* b_conv   = (const float*)d_in[11];
  const float* gWih     = (const float*)d_in[12];
  const float* gWhh     = (const float*)d_in[13];
  const float* gbih     = (const float*)d_in[14];
  const float* gbhh     = (const float*)d_in[15];
  const float* lWih     = (const float*)d_in[16];
  const float* lWhh     = (const float*)d_in[17];
  const float* lbih     = (const float*)d_in[18];
  const float* lbhh     = (const float*)d_in[19];

  const int N = in_sizes[0] / NDIM;
  const int E = in_sizes[2] / BDIM;

  // ---- workspace carve-up (256B aligned blocks) ----
  char* base = (char*)d_ws;
  size_t off = 0;
  auto alloc = [&](size_t bytes) -> char* {
    char* p = base + off;
    off = (off + bytes + 255) & ~(size_t)255;
    return p;
  };
  float*    out_b  = (float*)alloc((size_t)N * HD * 4);
  float*    aggr   = (float*)alloc((size_t)N * HD * 4);
  float*    cnt    = (float*)alloc((size_t)N * 4);
  _Float16* Af16   = (_Float16*)alloc((size_t)E * INTD * 2);
  _Float16* We2h   = (_Float16*)alloc((size_t)INTD * 4096 * 2);
  float*    e_ws   = (float*)alloc((size_t)N * 4);
  float*    a_ws   = (float*)alloc((size_t)N * 4);
  float*    small  = (float*)alloc((size_t)(NGRAPH + NGRAPH + NGRAPH * HD) * 4);
  unsigned* emax   = (unsigned*)small;
  float*    asum   = small + NGRAPH;
  float*    r_read = small + 2 * NGRAPH;
  float*    qstar  = (float*)alloc((size_t)NGRAPH * 2 * HD * 4);  // 8192 f
  float*    hlA    = (float*)alloc((size_t)NGRAPH * HD * 4);
  float*    hlB    = (float*)alloc((size_t)NGRAPH * HD * 4);
  float*    cl     = (float*)alloc((size_t)NGRAPH * HD * 4);
  _Float16* ew     = (_Float16*)alloc((size_t)E * 4096 * 2);      // 327 MB, last

  const int ZT = 256;
  // zero: qstar + hlA + hlB + cl are contiguous (each size a multiple of 256B)
  {
    int nz = NGRAPH * 2 * HD + 3 * NGRAPH * HD;  // 20480
    k_zero<<<(nz + ZT - 1) / ZT, ZT, 0, stream>>>(qstar, nz);
    k_zero<<<(N + ZT - 1) / ZT, ZT, 0, stream>>>(cnt, N);
  }

  // node expand (out == h), edge degree, edge MLP, We2 -> f16, big WMMA GEMM
  k_expand<<<N, HD, 0, stream>>>(x, W_expand, b_expand, out_b, N);
  k_count<<<(E + ZT - 1) / ZT, ZT, 0, stream>>>(ei, cnt, E);
  k_edge_mlp<<<E, INTD, 0, stream>>>(ea, We1, be1, Af16, E);
  k_cvt<<<(INTD * 4096 + ZT - 1) / ZT, ZT, 0, stream>>>(We2, We2h, INTD * 4096);
  {
    dim3 g((E + 127) / 128, 4096 / 64);
    k_ew_wmma<<<g, 256, 0, stream>>>(Af16, We2h, be2, ew, E);
  }

  // message passing + GRU, DEPTH iterations
  for (int d = 0; d < DEPTH; ++d) {
    k_zero<<<((N * HD) + ZT - 1) / ZT, ZT, 0, stream>>>(aggr, N * HD);
    k_msg<<<(E + 7) / 8, 256, 0, stream>>>(ei, ew, out_b, aggr, E);
    k_node<<<N, HD, 0, stream>>>(out_b, aggr, cnt, W_root, b_conv,
                                 gWih, gWhh, gbih, gbhh, N);
  }

  // Set2Set
  for (int s = 0; s < STEPS; ++s) {
    const float* hl_in = (s & 1) ? hlB : hlA;
    float*       hl_out = (s & 1) ? hlA : hlB;
    int nz = 2 * NGRAPH + NGRAPH * HD;  // emax + asum + r_read (contiguous)
    k_zero<<<(nz + ZT - 1) / ZT, ZT, 0, stream>>>(small, nz);
    k_lstm<<<(NGRAPH * HD) / 256, 256, 0, stream>>>(qstar, hl_in, cl, hl_out,
                                                    lWih, lWhh, lbih, lbhh);
    k_logits<<<(N + ZT - 1) / ZT, ZT, 0, stream>>>(out_b, hl_out, batch, e_ws, emax, N);
    k_attn<<<(N + ZT - 1) / ZT, ZT, 0, stream>>>(e_ws, emax, batch, a_ws, asum, N);
    k_read<<<((N * HD) + ZT - 1) / ZT, ZT, 0, stream>>>(a_ws, asum, batch, out_b,
                                                        r_read, N * HD);
    k_qstar<<<(NGRAPH * 2 * HD) / 256, 256, 0, stream>>>(hl_out, r_read, qstar);
  }

  k_copy<<<(out_size + ZT - 1) / ZT, ZT, 0, stream>>>(qstar, (float*)d_out, out_size);
}
